// Cross_TransformerModel_57561151701172
// MI455X (gfx1250) — compile-verified, matrix operand order checked by reading, not run on hardware
//
#include <hip/hip_runtime.h>
#include <hip/hip_bf16.h>
#include <cstdint>
#include <cstddef>

// ---------------------------------------------------------------------------
// Types for CDNA5 WMMA
// ---------------------------------------------------------------------------
typedef __bf16 v16bf __attribute__((ext_vector_type(16)));
typedef float  v8f   __attribute__((ext_vector_type(8)));
typedef unsigned short bf16_t;   // raw bf16 storage

__device__ __forceinline__ bf16_t f2bf(float f) {
  unsigned int u = __float_as_uint(f);
  u += 0x7FFFu + ((u >> 16) & 1u);      // round-to-nearest-even
  return (bf16_t)(u >> 16);
}

#define FLAG_RELU    1
#define FLAG_BF16OUT 2

// ---------------------------------------------------------------------------
// CDNA5 async global->LDS copy (GLOBAL_LOAD_ASYNC_TO_LDS_B128, ASYNCcnt).
// Probe result (round 2 stderr): builtin takes 4 args; param 1 is
// 'int __attribute__((vector_size(16))) __device__ *', i.e. non-const int4 in
// AS(1).  Param 2 is the LDS-side pointer (AS(3)); trailing args offset/cpol.
// Guarded: falls back to global_load_b128 + ds_store_b128 if absent.
// ---------------------------------------------------------------------------
#if defined(__has_builtin)
#if __has_builtin(__builtin_amdgcn_global_load_async_to_lds_b128)
#define HAVE_ASYNC_COPY 1
#endif
#endif

#ifdef HAVE_ASYNC_COPY
typedef int v4i_gcc __attribute__((vector_size(16)));
#define AS1_ __attribute__((address_space(1)))
#define AS3_ __attribute__((address_space(3)))

__device__ __forceinline__ void copy16_async(const bf16_t* g, bf16_t* l) {
  __builtin_amdgcn_global_load_async_to_lds_b128(
      (v4i_gcc AS1_*)g,
      (v4i_gcc AS3_*)l,
      0, 0);
}
__device__ __forceinline__ void wait_async0() {
#if __has_builtin(__builtin_amdgcn_s_wait_asynccnt)
  __builtin_amdgcn_s_wait_asynccnt(0);
#else
  asm volatile("s_wait_asynccnt 0x0" ::: "memory");
#endif
}
#endif

// ---------------------------------------------------------------------------
// WMMA GEMM:  C[M,N] = act( A[M,K](bf16) * B[N,K](bf16)^T + bias )
// Block tile 128x128, BLOCK_K = 32, 256 threads = 8 waves (4 in M x 2 in N),
// each wave computes 32x64 via 2x4 v_wmma_f32_16x16x32_bf16 tiles.
// LDS double-buffered, filled via async global->LDS when available.
// ---------------------------------------------------------------------------
__global__ __launch_bounds__(256)
void gemm_wmma_bf16(const bf16_t* __restrict__ A, int lda,
                    const bf16_t* __restrict__ B, int ldb,
                    const float*  __restrict__ bias,
                    void* __restrict__ C, int ldc,
                    int M, int N, int K, int flags)
{
  __shared__ __align__(16) bf16_t smA[2][128 * 32];
  __shared__ __align__(16) bf16_t smB[2][128 * 32];

  const int tid   = threadIdx.x;
  const int lane  = tid & 31;
  const int wave  = tid >> 5;      // 0..7
  const int waveM = wave & 3;      // 4 waves along M (32 rows each)
  const int waveN = wave >> 2;     // 2 waves along N (64 cols each)
  const int bm = blockIdx.y * 128;
  const int bn = blockIdx.x * 128;

  // A/B tiles: 128x32 bf16 -> 512 x 16B chunks each, 2 per thread per tile
  const int ca0 = tid,          ca1 = tid + 256;
  const int ra0 = ca0 >> 2,     ka0 = (ca0 & 3) * 8;
  const int ra1 = ca1 >> 2,     ka1 = (ca1 & 3) * 8;

  const int gra0 = min(bm + ra0, M - 1);
  const int gra1 = min(bm + ra1, M - 1);
  const int grb0 = min(bn + ra0, N - 1);
  const int grb1 = min(bn + ra1, N - 1);

  v8f acc[2][4];
  {
    v8f z = {0.f, 0.f, 0.f, 0.f, 0.f, 0.f, 0.f, 0.f};
#pragma unroll
    for (int tm = 0; tm < 2; ++tm)
#pragma unroll
      for (int tn = 0; tn < 4; ++tn) acc[tm][tn] = z;
  }

  const int nT = K >> 5;  // K is always a multiple of 32 here

  // ---- prologue: stage tile 0 into LDS buffer 0 ----
#ifdef HAVE_ASYNC_COPY
  copy16_async(A + (size_t)gra0 * lda + ka0, &smA[0][ra0 * 32 + ka0]);
  copy16_async(A + (size_t)gra1 * lda + ka1, &smA[0][ra1 * 32 + ka1]);
  copy16_async(B + (size_t)grb0 * ldb + ka0, &smB[0][ra0 * 32 + ka0]);
  copy16_async(B + (size_t)grb1 * ldb + ka1, &smB[0][ra1 * 32 + ka1]);
  wait_async0();
#else
  {
    uint4 a0 = *reinterpret_cast<const uint4*>(A + (size_t)gra0 * lda + ka0);
    uint4 a1 = *reinterpret_cast<const uint4*>(A + (size_t)gra1 * lda + ka1);
    uint4 b0 = *reinterpret_cast<const uint4*>(B + (size_t)grb0 * ldb + ka0);
    uint4 b1 = *reinterpret_cast<const uint4*>(B + (size_t)grb1 * ldb + ka1);
    *reinterpret_cast<uint4*>(&smA[0][ra0 * 32 + ka0]) = a0;
    *reinterpret_cast<uint4*>(&smA[0][ra1 * 32 + ka1]) = a1;
    *reinterpret_cast<uint4*>(&smB[0][ra0 * 32 + ka0]) = b0;
    *reinterpret_cast<uint4*>(&smB[0][ra1 * 32 + ka1]) = b1;
  }
#endif
  __syncthreads();

  const int kh = (lane >> 4) << 3;   // K-chunk base: 0 (lanes 0-15) or 8 (lanes 16-31)
  const int lr = lane & 15;

  int buf = 0;
  for (int t = 0; t < nT; ++t) {
    const bool hasNext = (t + 1) < nT;
    const int  nb = buf ^ 1;

#ifdef HAVE_ASYNC_COPY
    if (hasNext) {   // async engine fills the other buffer while we compute
      const int k0n = (t + 1) << 5;
      copy16_async(A + (size_t)gra0 * lda + k0n + ka0, &smA[nb][ra0 * 32 + ka0]);
      copy16_async(A + (size_t)gra1 * lda + k0n + ka1, &smA[nb][ra1 * 32 + ka1]);
      copy16_async(B + (size_t)grb0 * ldb + k0n + ka0, &smB[nb][ra0 * 32 + ka0]);
      copy16_async(B + (size_t)grb1 * ldb + k0n + ka1, &smB[nb][ra1 * 32 + ka1]);
    }
#else
    uint4 na0, na1, nb0, nb1;
    if (hasNext) {
      const int k0n = (t + 1) << 5;
      na0 = *reinterpret_cast<const uint4*>(A + (size_t)gra0 * lda + k0n + ka0);
      na1 = *reinterpret_cast<const uint4*>(A + (size_t)gra1 * lda + k0n + ka1);
      nb0 = *reinterpret_cast<const uint4*>(B + (size_t)grb0 * ldb + k0n + ka0);
      nb1 = *reinterpret_cast<const uint4*>(B + (size_t)grb1 * ldb + k0n + ka1);
      if ((t + 2) < nT) {
        const int k0nn = (t + 2) << 5;
        __builtin_prefetch(A + (size_t)gra0 * lda + k0nn + ka0, 0, 1);
        __builtin_prefetch(B + (size_t)grb0 * ldb + k0nn + ka0, 0, 1);
      }
    }
#endif

    // Build fragments from LDS (two ds_load_b128 per fragment; CDNA5 16-bit
    // layout: VGPR0-3 hold K = kh..kh+7, VGPR4-7 hold K = kh+16..kh+23).
    const bf16_t* As = smA[buf];
    const bf16_t* Bs = smB[buf];
    v16bf af[2], bfb[4];
#pragma unroll
    for (int tm = 0; tm < 2; ++tm) {
      const bf16_t* src = As + (waveM * 32 + tm * 16 + lr) * 32 + kh;
      uint4* d = reinterpret_cast<uint4*>(&af[tm]);
      d[0] = *reinterpret_cast<const uint4*>(src);
      d[1] = *reinterpret_cast<const uint4*>(src + 16);
    }
#pragma unroll
    for (int tn = 0; tn < 4; ++tn) {
      const bf16_t* src = Bs + (waveN * 64 + tn * 16 + lr) * 32 + kh;
      uint4* d = reinterpret_cast<uint4*>(&bfb[tn]);
      d[0] = *reinterpret_cast<const uint4*>(src);
      d[1] = *reinterpret_cast<const uint4*>(src + 16);
    }

#pragma unroll
    for (int tm = 0; tm < 2; ++tm)
#pragma unroll
      for (int tn = 0; tn < 4; ++tn)
        acc[tm][tn] = __builtin_amdgcn_wmma_f32_16x16x32_bf16(
            /*neg_a=*/false, af[tm], /*neg_b=*/false, bfb[tn],
            /*c_mod=*/(short)0, acc[tm][tn],
            /*reuse_a=*/false, /*reuse_b=*/false);

#ifdef HAVE_ASYNC_COPY
    if (hasNext) wait_async0();
#else
    if (hasNext) {
      *reinterpret_cast<uint4*>(&smA[nb][ra0 * 32 + ka0]) = na0;
      *reinterpret_cast<uint4*>(&smA[nb][ra1 * 32 + ka1]) = na1;
      *reinterpret_cast<uint4*>(&smB[nb][ra0 * 32 + ka0]) = nb0;
      *reinterpret_cast<uint4*>(&smB[nb][ra1 * 32 + ka1]) = nb1;
    }
#endif
    __syncthreads();
    buf = nb;
  }

  // Epilogue: C layout = 8 VGPRs, lanes 0-15 rows i, lanes 16-31 rows 8+i.
#pragma unroll
  for (int tm = 0; tm < 2; ++tm) {
#pragma unroll
    for (int tn = 0; tn < 4; ++tn) {
      const int row0 = bm + waveM * 32 + tm * 16 + ((lane >> 4) << 3);
      const int col  = bn + waveN * 64 + tn * 16 + (lane & 15);
      const float bb = (bias != nullptr && col < N) ? bias[col] : 0.f;
#pragma unroll
      for (int i = 0; i < 8; ++i) {
        const int row = row0 + i;
        float v = acc[tm][tn][i] + bb;
        if (flags & FLAG_RELU) v = fmaxf(v, 0.f);
        if (row < M && col < N) {
          if (flags & FLAG_BF16OUT)
            reinterpret_cast<bf16_t*>(C)[(size_t)row * ldc + col] = f2bf(v);
          else
            reinterpret_cast<float*>(C)[(size_t)row * ldc + col] = v;
        }
      }
    }
  }
}

// ---------------------------------------------------------------------------
// Row softmax over Sv=196 cols (one wave32 per row), writes bf16 P padded to
// SvP=224 with zeros.  s = (scores + add) * scale.
// ---------------------------------------------------------------------------
__global__ __launch_bounds__(32)
void softmax_kernel(const float* __restrict__ S, int ldS,
                    const float* __restrict__ addm, int Sv, int SvP,
                    bf16_t* __restrict__ P, float scale)
{
  const int row  = blockIdx.x;
  const int lane = threadIdx.x;
  float v[7];
  float m = -3.0e38f;
#pragma unroll
  for (int t = 0; t < 7; ++t) {
    const int j = lane + t * 32;
    float x = -3.0e38f;
    if (j < Sv) x = (S[(size_t)row * ldS + j] + addm[(size_t)row * Sv + j]) * scale;
    v[t] = x;
    m = fmaxf(m, x);
  }
#pragma unroll
  for (int o = 16; o > 0; o >>= 1) m = fmaxf(m, __shfl_xor(m, o, 32));
  float s = 0.f;
#pragma unroll
  for (int t = 0; t < 7; ++t) {
    const int j = lane + t * 32;
    const float e = (j < Sv) ? __expf(v[t] - m) : 0.f;
    v[t] = e;
    s += e;
  }
#pragma unroll
  for (int o = 16; o > 0; o >>= 1) s += __shfl_xor(s, o, 32);
  const float inv = 1.f / s;
#pragma unroll
  for (int t = 0; t < 7; ++t) {
    const int j = lane + t * 32;
    if (j < SvP)
      P[(size_t)row * SvP + j] = (j < Sv) ? f2bf(v[t] * inv) : (bf16_t)0;
  }
}

// ---------------------------------------------------------------------------
// Fused residual + LayerNorm over D=1024, one block (256 thr) per row.
// ---------------------------------------------------------------------------
__global__ __launch_bounds__(256)
void add_layernorm(const float* __restrict__ X, const float* __restrict__ R,
                   const float* __restrict__ g, const float* __restrict__ b,
                   float* __restrict__ out)
{
  __shared__ float rowbuf[1024];
  __shared__ float redx[256];
  __shared__ float redy[256];
  const int row = blockIdx.x;
  const int tid = threadIdx.x;
  float s = 0.f, ss = 0.f;
#pragma unroll
  for (int i = 0; i < 4; ++i) {
    const int j = tid + i * 256;
    const float v = X[(size_t)row * 1024 + j] + R[(size_t)row * 1024 + j];
    rowbuf[j] = v;
    s += v;
    ss += v * v;
  }
  redx[tid] = s;
  redy[tid] = ss;
  __syncthreads();
  for (int o = 128; o > 0; o >>= 1) {
    if (tid < o) { redx[tid] += redx[tid + o]; redy[tid] += redy[tid + o]; }
    __syncthreads();
  }
  const float mu  = redx[0] * (1.f / 1024.f);
  const float var = redy[0] * (1.f / 1024.f) - mu * mu;
  const float rs  = rsqrtf(var + 1e-6f);
#pragma unroll
  for (int i = 0; i < 4; ++i) {
    const int j = tid + i * 256;
    out[(size_t)row * 1024 + j] = (rowbuf[j] - mu) * rs * g[j] + b[j];
  }
}

// ---------------------------------------------------------------------------
// Concat(out1,out2) + MaxPool1d(k=2,s=2):  pooled[:, :512] = pool(out1),
// pooled[:, 512:] = pool(out2).  Emits fp32 and bf16 copies.
// ---------------------------------------------------------------------------
__global__ __launch_bounds__(256)
void pool_kernel(const float* __restrict__ o1, const float* __restrict__ o2,
                 float* __restrict__ pf, bf16_t* __restrict__ pb, int rows)
{
  const int idx = blockIdx.x * 256 + threadIdx.x;
  if (idx >= rows * 1024) return;
  const int r = idx >> 10;
  const int m = idx & 1023;
  float v;
  if (m < 512) {
    v = fmaxf(o1[(size_t)r * 1024 + 2 * m], o1[(size_t)r * 1024 + 2 * m + 1]);
  } else {
    const int mm = m - 512;
    v = fmaxf(o2[(size_t)r * 1024 + 2 * mm], o2[(size_t)r * 1024 + 2 * mm + 1]);
  }
  pf[idx] = v;
  pb[idx] = f2bf(v);
}

// ---------------------------------------------------------------------------
// fp32 -> bf16 elementwise (grid-stride)
// ---------------------------------------------------------------------------
__global__ __launch_bounds__(256)
void cvt_f32_to_bf16(const float* __restrict__ in, bf16_t* __restrict__ out, long n)
{
  long i = (long)blockIdx.x * blockDim.x + threadIdx.x;
  const long stride = (long)gridDim.x * blockDim.x;
  for (; i < n; i += stride) out[i] = f2bf(in[i]);
}

// ---------------------------------------------------------------------------
// Vt[n, k] = V[k, n] as bf16, zero-padded for k in [Sv, SvP)
// ---------------------------------------------------------------------------
__global__ __launch_bounds__(256)
void transpose_vt(const float* __restrict__ V, bf16_t* __restrict__ Vt,
                  int D, int Sv, int SvP)
{
  const int idx = blockIdx.x * 256 + threadIdx.x;
  if (idx >= D * SvP) return;
  const int n = idx / SvP;
  const int k = idx - n * SvP;
  Vt[idx] = (k < Sv) ? f2bf(V[(size_t)k * D + n]) : (bf16_t)0;
}

// ---------------------------------------------------------------------------
// Host orchestration
// ---------------------------------------------------------------------------
extern "C" void kernel_launch(void* const* d_in, const int* in_sizes, int n_in,
                              void* d_out, int out_size, void* d_ws, size_t ws_size,
                              hipStream_t stream)
{
  (void)in_sizes; (void)n_in; (void)out_size; (void)ws_size;
  const int Bb = 4, Sq = 2048, Sv = 196, SvP = 224, D = 1024, DFF = 4096;
  const int Mrows = Bb * Sq;  // 8192

  const float* text1 = (const float*)d_in[0];
  const float* kv1   = (const float*)d_in[1];
  const float* sc1   = (const float*)d_in[2];
  const float* text2 = (const float*)d_in[3];
  const float* kv2   = (const float*)d_in[4];
  const float* sc2   = (const float*)d_in[5];
  const float* Wq = (const float*)d_in[6];   const float* bq  = (const float*)d_in[7];
  const float* Wk = (const float*)d_in[8];   const float* bk  = (const float*)d_in[9];
  const float* Wv = (const float*)d_in[10];  const float* bv  = (const float*)d_in[11];
  const float* Wo = (const float*)d_in[12];  const float* bo  = (const float*)d_in[13];
  const float* W1 = (const float*)d_in[14];  const float* b1  = (const float*)d_in[15];
  const float* W2 = (const float*)d_in[16];  const float* b2  = (const float*)d_in[17];
  const float* g1 = (const float*)d_in[18];  const float* be1 = (const float*)d_in[19];
  const float* g2 = (const float*)d_in[20];  const float* be2 = (const float*)d_in[21];

  char* ws = (char*)d_ws;
  size_t off = 0;
  auto alloc = [&](size_t bytes) -> char* {
    char* p = ws + off;
    off = (off + bytes + 255) & ~(size_t)255;
    return p;
  };

  // Persistent region
  bf16_t* wq_bf = (bf16_t*)alloc((size_t)D * D * 2);
  bf16_t* wk_bf = (bf16_t*)alloc((size_t)D * D * 2);
  bf16_t* wv_bf = (bf16_t*)alloc((size_t)D * D * 2);
  bf16_t* wo_bf = (bf16_t*)alloc((size_t)D * D * 2);
  bf16_t* w1_bf = (bf16_t*)alloc((size_t)DFF * D * 2);
  bf16_t* w2_bf = (bf16_t*)alloc((size_t)D * DFF * 2);
  float*  out1  = (float*)alloc((size_t)Mrows * D * 4);
  float*  out2  = (float*)alloc((size_t)Mrows * D * 4);

  // Branch-phase scratch (reused for both branches)
  const size_t region = off;
  bf16_t* Xbf  = (bf16_t*)alloc((size_t)Mrows * D * 2);
  bf16_t* Qbf  = (bf16_t*)alloc((size_t)Mrows * D * 2);
  bf16_t* ATb  = (bf16_t*)alloc((size_t)Mrows * D * 2);
  float*  Sco  = (float*)alloc((size_t)Mrows * SvP * 4);
  bf16_t* Pbf  = (bf16_t*)alloc((size_t)Mrows * SvP * 2);
  bf16_t* KVbf = (bf16_t*)alloc((size_t)Sv * D * 2);
  bf16_t* Kbf  = (bf16_t*)alloc((size_t)Sv * D * 2);
  float*  Vf   = (float*)alloc((size_t)Sv * D * 4);
  bf16_t* Vtb  = (bf16_t*)alloc((size_t)D * SvP * 2);
  float*  Of   = (float*)alloc((size_t)Mrows * D * 4);

  // FFN-phase scratch overlaps branch-phase scratch (branch scratch is dead by then)
  off = region;
  float*  pooledF = (float*)alloc((size_t)Mrows * D * 4);
  bf16_t* pooledB = (bf16_t*)alloc((size_t)Mrows * D * 2);
  bf16_t* Hbf     = (bf16_t*)alloc((size_t)Mrows * DFF * 2);
  float*  Ff      = (float*)alloc((size_t)Mrows * D * 4);

  auto cvt = [&](const float* src, bf16_t* dst, long n) {
    long blocks = (n + 255) / 256;
    if (blocks > 4096) blocks = 4096;
    cvt_f32_to_bf16<<<dim3((unsigned)blocks), dim3(256), 0, stream>>>(src, dst, n);
  };
  auto gemm = [&](const bf16_t* A, int lda, const bf16_t* Bm, int ldb,
                  const float* bias, void* C, int ldc,
                  int M, int N, int K, int flags) {
    dim3 grid((N + 127) / 128, (M + 127) / 128);
    gemm_wmma_bf16<<<grid, dim3(256), 0, stream>>>(A, lda, Bm, ldb, bias, C, ldc,
                                                   M, N, K, flags);
  };

  // Weight conversion (shared by both branches)
  cvt(Wq, wq_bf, (long)D * D);
  cvt(Wk, wk_bf, (long)D * D);
  cvt(Wv, wv_bf, (long)D * D);
  cvt(Wo, wo_bf, (long)D * D);
  cvt(W1, w1_bf, (long)DFF * D);
  cvt(W2, w2_bf, (long)D * DFF);

  for (int br = 0; br < 2; ++br) {
    const float* X  = br ? text2 : text1;
    const float* kv = br ? kv2 : kv1;
    const float* sc = br ? sc2 : sc1;
    float* outb = br ? out2 : out1;

    cvt(X, Xbf, (long)Mrows * D);
    cvt(kv, KVbf, (long)Sv * D);

    // K = kv @ Wk^T + bk  (bf16 out, row-major [Sv, D] == B[N,K] for scores GEMM)
    gemm(KVbf, D, wk_bf, D, bk, Kbf, D, Sv, D, D, FLAG_BF16OUT);
    // V = kv @ Wv^T + bv  (fp32), then transpose to Vt[N=D, K=SvP] bf16
    gemm(KVbf, D, wv_bf, D, bv, Vf, D, Sv, D, D, 0);
    transpose_vt<<<dim3((D * SvP + 255) / 256), dim3(256), 0, stream>>>(
        Vf, Vtb, D, Sv, SvP);
    // Q = X @ Wq^T + bq (bf16 out)
    gemm(Xbf, D, wq_bf, D, bq, Qbf, D, Mrows, D, D, FLAG_BF16OUT);
    // scores = Q @ K^T  (fp32, ldc = SvP)
    gemm(Qbf, D, Kbf, D, nullptr, Sco, SvP, Mrows, Sv, D, 0);
    // softmax((scores + score_matrix) / 8) -> bf16 P, zero-padded to SvP
    softmax_kernel<<<dim3(Mrows), dim3(32), 0, stream>>>(
        Sco, SvP, sc, Sv, SvP, Pbf, 0.125f);
    // attn = P @ V  (bf16 out)
    gemm(Pbf, SvP, Vtb, SvP, nullptr, ATb, D, Mrows, D, SvP, FLAG_BF16OUT);
    // O = attn @ Wo^T + bo  (fp32)
    gemm(ATb, D, wo_bf, D, bo, Of, D, Mrows, D, D, 0);
    // out_i = LN(X + O)
    add_layernorm<<<dim3(Mrows), dim3(256), 0, stream>>>(X, Of, g1, be1, outb);
  }

  // concat + maxpool
  pool_kernel<<<dim3((Mrows * D + 255) / 256), dim3(256), 0, stream>>>(
      out1, out2, pooledF, pooledB, Mrows);
  // FFN: H = relu(pooled @ W1^T + b1)  (bf16 out)
  gemm(pooledB, D, w1_bf, D, b1, Hbf, DFF, Mrows, DFF, D, FLAG_RELU | FLAG_BF16OUT);
  // F = H @ W2^T + b2  (fp32)
  gemm(Hbf, DFF, w2_bf, DFF, b2, Ff, D, Mrows, D, DFF, 0);
  // out = LN(pooled + F)
  add_layernorm<<<dim3(Mrows), dim3(256), 0, stream>>>(
      pooledF, Ff, g2, be2, (float*)d_out);
}
